// SelfAttn_2087354105932
// MI455X (gfx1250) — matrix operand order repaired and checked by reading, not compiled, not run on hardware
//
#include <hip/hip_runtime.h>
#include <hip/hip_bf16.h>

// ---------------------------------------------------------------------------
// Self-attention block (LN -> QKV -> MHA(softmax) -> proj) for MI455X/gfx1250.
// All GEMMs on v_wmma_f32_16x16x32_bf16 (wave32). fp32 -> bf16 A/B, f32 accum.
// Problem is tiny vs the chip (52 GFLOP, ~35 MB working set < 192 MB L2):
//  - GEMM fragments load straight from global (b128, L2-resident).
//  - Attention is fused flash-style; K/V tiles are staged into LDS once per
//    block with gfx1250 async global->LDS copies (ASYNCcnt), double-buffered,
//    cutting redundant per-wave L2 reads 8x (~2.1 GB -> ~0.26 GB).
// ---------------------------------------------------------------------------

#define DIMC 512
#define SEQ  2048
#define NBAT 4
#define NH   8
#define HD   64
#define NT   (NBAT * SEQ)   // 8192 tokens

typedef __attribute__((ext_vector_type(16))) __bf16 v16bf;
typedef __attribute__((ext_vector_type(8)))  __bf16 v8bf;
typedef __attribute__((ext_vector_type(8)))  float  v8f;

union Frag { v16bf v; v8bf h[2]; };

static __device__ __forceinline__ v8f wmma_bf16(v16bf a, v16bf b, v8f c) {
  // D = A(16x32 bf16) * B(32x16 bf16) + C(16x16 f32)
  return __builtin_amdgcn_wmma_f32_16x16x32_bf16(false, a, false, b, (short)0, c,
                                                 false, false);
}

// A fragment: lane L holds row (L&15); halves split the K dim:
//   elems 0..7 at k = 8*half, elems 8..15 at k = 16 + 8*half   (k-contiguous)
static __device__ __forceinline__ Frag load_a(const __bf16* rowPtr, int half) {
  Frag f;
  f.h[0] = *(const v8bf*)(rowPtr + 8 * half);
  f.h[1] = *(const v8bf*)(rowPtr + 16 + 8 * half);
  return f;
}
// B fragment: lane L holds col (L&15); elems 0..7 at k = 8*half, elems 8..15 at k = 16 + 8*half
static __device__ __forceinline__ Frag load_b(const __bf16* colPtr, int half) {
  Frag f;
  f.h[0] = *(const v8bf*)(colPtr + 8 * half);
  f.h[1] = *(const v8bf*)(colPtr + 16 + 8 * half);
  return f;
}

// gfx1250 async global->LDS copy of 16 bytes per lane (tracked by ASYNCcnt).
// Flat addresses to LDS keep the LDS byte offset in addr[31:0].
static __device__ __forceinline__ void async_b128(void* lds, const void* g) {
  unsigned loff = (unsigned)(uintptr_t)lds;
  asm volatile("global_load_async_to_lds_b128 %0, %1, off"
               :: "v"(loff), "v"(g) : "memory");
}
static __device__ __forceinline__ void wait_async0() {
  asm volatile("s_wait_asynccnt 0x0" ::: "memory");
}
static __device__ __forceinline__ void wait_ds0() {
  asm volatile("s_wait_dscnt 0x0" ::: "memory");
}

// ---------------------------------------------------------------------------
// K1: LayerNorm over dim, fp32 (n,dim,s) -> bf16 (n*s, dim). Coalesced on s.
// ---------------------------------------------------------------------------
__global__ __launch_bounds__(256) void k_ln(const float* __restrict__ x,
                                            const float* __restrict__ g,
                                            const float* __restrict__ b,
                                            __bf16* __restrict__ xn) {
  int t = blockIdx.x * 256 + threadIdx.x;           // token id, 0..8191
  int n = t >> 11, sIdx = t & (SEQ - 1);
  const float* xp = x + (size_t)n * DIMC * SEQ + sIdx;
  float sum = 0.f, sq = 0.f;
  for (int d = 0; d < DIMC; ++d) {
    float v = xp[(size_t)d * SEQ];
    sum += v; sq += v * v;
  }
  float mean = sum * (1.0f / DIMC);
  float var  = sq  * (1.0f / DIMC) - mean * mean;
  float rstd = rsqrtf(var + 1e-5f);
  __bf16* op = xn + (size_t)t * DIMC;
  for (int d = 0; d < DIMC; ++d)
    op[d] = (__bf16)(((xp[(size_t)d * SEQ] - mean) * rstd) * g[d] + b[d]);
}

// ---------------------------------------------------------------------------
// KW: weight convert+transpose: w[K][C] f32 -> wT[C][K] bf16 (K-contiguous so
// B fragments load straight from global).
// ---------------------------------------------------------------------------
__global__ __launch_bounds__(256) void k_wcvt(const float* __restrict__ w,
                                              __bf16* __restrict__ wt,
                                              int K, int C) {
  int i = blockIdx.x * 256 + threadIdx.x;
  if (i < K * C) {
    int k = i / C, c = i % C;
    wt[(size_t)c * K + k] = (__bf16)w[i];
  }
}

// ---------------------------------------------------------------------------
// K2: QKV GEMM  (8192 x 512) x (512 x 1536) + bias, scatter into
//   q [nh][s][64], k [nh][s][64], vT [nh][64][s]  (all bf16).
// Block = 64 rows x 128 cols; wave = 16 rows x 64 cols (4 WMMA tiles, A reuse).
// ---------------------------------------------------------------------------
__global__ __launch_bounds__(256) void k_qkv(const __bf16* __restrict__ xn,
                                             const __bf16* __restrict__ wqkvT,
                                             const float* __restrict__ bqkv,
                                             __bf16* __restrict__ q,
                                             __bf16* __restrict__ kk,
                                             __bf16* __restrict__ vT) {
  int lane = threadIdx.x & 31, wave = threadIdx.x >> 5;
  int half = lane >> 4, ln16 = lane & 15;
  int rowBase = blockIdx.x * 64 + (wave & 3) * 16;
  int colBase = blockIdx.y * 128 + (wave >> 2) * 64;

  v8f z = {0.f, 0.f, 0.f, 0.f, 0.f, 0.f, 0.f, 0.f};
  v8f acc[4] = {z, z, z, z};

  const __bf16* aRow = xn + (size_t)(rowBase + ln16) * DIMC;
  for (int k0 = 0; k0 < DIMC; k0 += 32) {
    Frag A = load_a(aRow + k0, half);
#pragma unroll
    for (int t = 0; t < 4; ++t) {
      Frag B = load_b(wqkvT + (size_t)(colBase + t * 16 + ln16) * DIMC + k0, half);
      acc[t] = wmma_bf16(A.v, B.v, acc[t]);
    }
  }
#pragma unroll
  for (int t = 0; t < 4; ++t) {
    int gCol  = colBase + t * 16 + ln16;          // 0..1535
    float bia = bqkv[gCol];
    int hd = gCol / 192, inner = gCol % 192;      // (h, 3*64) split
#pragma unroll
    for (int r = 0; r < 8; ++r) {
      int gRow = rowBase + r + 8 * half;
      int n = gRow >> 11, sIdx = gRow & (SEQ - 1);
      __bf16 bv = (__bf16)(acc[t][r] + bia);
      size_t nh = (size_t)(n * NH + hd);
      if (inner < 64)        q [(nh * SEQ + sIdx) * HD + inner]        = bv;
      else if (inner < 128)  kk[(nh * SEQ + sIdx) * HD + (inner - 64)] = bv;
      else                   vT[(nh * HD + (inner - 128)) * SEQ + sIdx] = bv;
    }
  }
}

// ---------------------------------------------------------------------------
// K3: fused flash attention per (n,h). Block = 128 queries (wave = 16).
// K/V tiles (32 keys) staged into LDS once per block via async global->LDS
// copies, double-buffered. Per tile: 4 WMMA for S = Q K^T, online softmax
// (shfl reductions in each 16-lane half), P transposed C->A layout through
// per-wave LDS, then 4 WMMA for O += P V.
// ---------------------------------------------------------------------------
#define KPITCH 72   // halves; 36 dwords -> conflict-free across 16 lanes
#define VPITCH 40   // halves; 20 dwords -> conflict-free across 16 lanes

__global__ __launch_bounds__(256) void k_attn(const __bf16* __restrict__ q,
                                              const __bf16* __restrict__ kk,
                                              const __bf16* __restrict__ vT,
                                              __bf16* __restrict__ attn) {
  __shared__ alignas(16) __bf16 kt[2][32 * KPITCH];   // [buf][token][d]
  __shared__ alignas(16) __bf16 vt[2][64 * VPITCH];   // [buf][d][token]
  __shared__ alignas(16) __bf16 lds_p[8][16 * 40];    // per-wave P tile

  int tid = threadIdx.x;
  int lane = tid & 31, wave = tid >> 5;
  int half = lane >> 4, ln16 = lane & 15;
  int nh = blockIdx.x;                            // n*8 + h
  int n = nh >> 3, hd = nh & 7;
  int qBase = blockIdx.y * 128 + wave * 16;

  const __bf16* qhp = q  + (size_t)nh * SEQ * HD;
  const __bf16* khp = kk + (size_t)nh * SEQ * HD;
  const __bf16* vhp = vT + (size_t)nh * HD * SEQ;

  // async-stage one 32-key K tile and V tile into LDS buffer `buf`
  auto stage = [&](int buf, int kb) {
    int krow = tid >> 3, kch = tid & 7;           // 32 rows x 8 chunks of 8
    async_b128(&kt[buf][krow * KPITCH + kch * 8],
               khp + (size_t)(kb + krow) * HD + kch * 8);
    int vrow = tid >> 2, vch = tid & 3;           // 64 rows x 4 chunks of 8
    async_b128(&vt[buf][vrow * VPITCH + vch * 8],
               vhp + (size_t)vrow * SEQ + kb + vch * 8);
  };

  stage(0, 0);

  // Q fragments for d-chunks [0,32) and [32,64): resident all loop long.
  const __bf16* qRow = qhp + (size_t)(qBase + ln16) * HD;
  Frag A0 = load_a(qRow, half);
  Frag A1 = load_a(qRow + 32, half);

  v8f z = {0.f, 0.f, 0.f, 0.f, 0.f, 0.f, 0.f, 0.f};
  v8f O[4] = {z, z, z, z};
  float m[8], l[8];
#pragma unroll
  for (int r = 0; r < 8; ++r) { m[r] = -3.0e38f; l[r] = 0.f; }

  wait_async0();
  __syncthreads();

  int buf = 0;
  for (int kb = 0; kb < SEQ; kb += 32) {
    if (kb + 32 < SEQ) stage(buf ^ 1, kb + 32);   // prefetch next tile (async)

    // ---- S = Q K^T for two 16-key sub-tiles (from LDS) ---------------------
    v8f s0 = z, s1 = z;
    {
      const __bf16* k0p = &kt[buf][ln16 * KPITCH];
      const __bf16* k1p = &kt[buf][(16 + ln16) * KPITCH];
      Frag B;
      B = load_b(k0p, half);       s0 = wmma_bf16(A0.v, B.v, s0);
      B = load_b(k0p + 32, half);  s0 = wmma_bf16(A1.v, B.v, s0);
      B = load_b(k1p, half);       s1 = wmma_bf16(A0.v, B.v, s1);
      B = load_b(k1p + 32, half);  s1 = wmma_bf16(A1.v, B.v, s1);
    }

    // ---- online softmax (rows live across the 16 lanes of this half) ------
    float p0[8], p1[8], alpha[8];
#pragma unroll
    for (int r = 0; r < 8; ++r) {
      float a = s0[r] * 0.125f, b = s1[r] * 0.125f;   // 1/SCALE = 1/8
      float tm = fmaxf(a, b);
#pragma unroll
      for (int msk = 1; msk < 16; msk <<= 1) tm = fmaxf(tm, __shfl_xor(tm, msk, 32));
      float mn = fmaxf(m[r], tm);
      alpha[r] = __expf(m[r] - mn);
      m[r] = mn;
      p0[r] = __expf(a - mn);
      p1[r] = __expf(b - mn);
      float ps = p0[r] + p1[r];
#pragma unroll
      for (int msk = 1; msk < 16; msk <<= 1) ps += __shfl_xor(ps, msk, 32);
      l[r] = l[r] * alpha[r] + ps;
    }
#pragma unroll
    for (int t = 0; t < 4; ++t)
#pragma unroll
      for (int r = 0; r < 8; ++r) O[t][r] *= alpha[r];

    // ---- transpose P from C-layout(f32) to A-layout(bf16) via wave LDS ----
    __bf16* pw = lds_p[wave];
#pragma unroll
    for (int r = 0; r < 8; ++r) {
      int row = r + 8 * half;
      pw[row * 40 + ln16]      = (__bf16)p0[r];
      pw[row * 40 + 16 + ln16] = (__bf16)p1[r];
    }
    wait_ds0();                                    // wave-local LDS RAW fence
    Frag P = load_a(pw + ln16 * 40, half);

    // ---- O += P V  (V tile in LDS, token index contiguous) ----------------
#pragma unroll
    for (int t = 0; t < 4; ++t) {
      Frag BV = load_b(&vt[buf][(t * 16 + ln16) * VPITCH], half);
      O[t] = wmma_bf16(P.v, BV.v, O[t]);
    }

    // close WAR window (outstanding ds reads) then sync tile hand-off
    wait_ds0();
    wait_async0();
    __syncthreads();
    buf ^= 1;
  }

  // ---- normalize and store O as attn[n][s][h*64+d] (bf16 for proj GEMM) ---
  float inv[8];
#pragma unroll
  for (int r = 0; r < 8; ++r) inv[r] = 1.0f / l[r];
#pragma unroll
  for (int t = 0; t < 4; ++t)
#pragma unroll
    for (int r = 0; r < 8; ++r) {
      int sRow = qBase + r + 8 * half;
      attn[((size_t)n * SEQ + sRow) * DIMC + hd * HD + t * 16 + ln16] =
          (__bf16)(O[t][r] * inv[r]);
    }
}

// ---------------------------------------------------------------------------
// K4: output projection (8192 x 512) x (512 x 512) + bias, storing fp32
// directly in the transposed (n, dim, s) output layout.
// ---------------------------------------------------------------------------
__global__ __launch_bounds__(256) void k_proj(const __bf16* __restrict__ attn,
                                              const __bf16* __restrict__ wprojT,
                                              const float* __restrict__ bproj,
                                              float* __restrict__ out) {
  int lane = threadIdx.x & 31, wave = threadIdx.x >> 5;
  int half = lane >> 4, ln16 = lane & 15;
  int rowBase = blockIdx.x * 64 + (wave & 3) * 16;
  int colBase = blockIdx.y * 128 + (wave >> 2) * 64;

  v8f z = {0.f, 0.f, 0.f, 0.f, 0.f, 0.f, 0.f, 0.f};
  v8f acc[4] = {z, z, z, z};

  const __bf16* aRow = attn + (size_t)(rowBase + ln16) * DIMC;
  for (int k0 = 0; k0 < DIMC; k0 += 32) {
    Frag A = load_a(aRow + k0, half);
#pragma unroll
    for (int t = 0; t < 4; ++t) {
      Frag B = load_b(wprojT + (size_t)(colBase + t * 16 + ln16) * DIMC + k0, half);
      acc[t] = wmma_bf16(A.v, B.v, acc[t]);
    }
  }
#pragma unroll
  for (int t = 0; t < 4; ++t) {
    int gCol = colBase + t * 16 + ln16;
    float bia = bproj[gCol];
#pragma unroll
    for (int r = 0; r < 8; ++r) {
      int gRow = rowBase + r + 8 * half;
      int n = gRow >> 11, sIdx = gRow & (SEQ - 1);
      out[(size_t)n * DIMC * SEQ + (size_t)gCol * SEQ + sIdx] = acc[t][r] + bia;
    }
  }
}

// ---------------------------------------------------------------------------
extern "C" void kernel_launch(void* const* d_in, const int* in_sizes, int n_in,
                              void* d_out, int out_size, void* d_ws, size_t ws_size,
                              hipStream_t stream) {
  const float* x      = (const float*)d_in[0];
  const float* ln_g   = (const float*)d_in[1];
  const float* ln_b   = (const float*)d_in[2];
  const float* w_qkv  = (const float*)d_in[3];
  const float* b_qkv  = (const float*)d_in[4];
  const float* w_proj = (const float*)d_in[5];
  const float* b_proj = (const float*)d_in[6];
  float* out = (float*)d_out;

  char* ws = (char*)d_ws;
  // workspace layout (bytes); xn buffer is reused for the attention output
  __bf16* xn     = (__bf16*)(ws);                         // 8192*512   (8.39 MB)
  __bf16* qbuf   = (__bf16*)(ws + 8388608ull);            // 32*2048*64 (8.39 MB)
  __bf16* kbuf   = (__bf16*)(ws + 16777216ull);
  __bf16* vTbuf  = (__bf16*)(ws + 25165824ull);
  __bf16* wqkvT  = (__bf16*)(ws + 33554432ull);           // 1536*512 bf16
  __bf16* wprojT = (__bf16*)(ws + 35127296ull);           // 512*512 bf16
  __bf16* attn   = xn;                                    // reuse after K2

  // weights -> bf16, transposed (K-contiguous)
  k_wcvt<<<dim3((DIMC * 3 * DIMC + 255) / 256), dim3(256), 0, stream>>>(
      w_qkv, wqkvT, DIMC, 3 * DIMC);
  k_wcvt<<<dim3((DIMC * DIMC + 255) / 256), dim3(256), 0, stream>>>(
      w_proj, wprojT, DIMC, DIMC);

  // layernorm + bf16
  k_ln<<<dim3(NT / 256), dim3(256), 0, stream>>>(x, ln_g, ln_b, xn);

  // qkv projection
  k_qkv<<<dim3(NT / 64, (3 * DIMC) / 128), dim3(256), 0, stream>>>(
      xn, wqkvT, b_qkv, qbuf, kbuf, vTbuf);

  // fused flash attention
  k_attn<<<dim3(NBAT * NH, SEQ / 128), dim3(256), 0, stream>>>(
      qbuf, kbuf, vTbuf, attn);

  // output projection (+ transpose to (n, dim, s))
  k_proj<<<dim3(NT / 64, DIMC / 128), dim3(256), 0, stream>>>(
      attn, wprojT, b_proj, out);
}
